// VAE_GNN_59304908423330
// MI455X (gfx1250) — matrix operand order repaired and checked by reading, not compiled, not run on hardware
//
#include <hip/hip_runtime.h>
#include <cstdint>
#include <cstddef>

typedef __attribute__((ext_vector_type(16))) __bf16 v16bf;
typedef __attribute__((ext_vector_type(8)))  __bf16 v8bf;
typedef __attribute__((ext_vector_type(8)))  float  v8f;

#define NN  10000
#define NNP 10048          // padded rows: multiple of 64 (157 * 64)
#define NE  160000

// ---------------------------------------------------------------------------
// Register-blocked bf16 WMMA GEMM:  C[N,O] = A[NNP,KP] * B[OP,KP]^T (+bias)(+leaky)
// One wave computes a 64x32 output block = 4x2 grid of 16x16 WMMA tiles.
// A has NNP rows (zero padded), B has ceil32(O) rows (zero padded), K padded
// to KP (multiple of 32). mode: 0=linear, 1=+bias, 2=+bias+leaky.
// ---------------------------------------------------------------------------
__global__ __launch_bounds__(256)
void k_gemm_bf16(const __bf16* __restrict__ A, const __bf16* __restrict__ B,
                 const float* __restrict__ bias, float* __restrict__ C,
                 int KP, int O, int ldc, int CT2, int totalWaves,
                 int mode, float slope)
{
    int wave = blockIdx.x * 8 + (threadIdx.x >> 5);
    if (wave >= totalWaves) return;            // uniform per wave -> EXEC all-1s below
    int lane = threadIdx.x & 31;
    int rt = wave / CT2;
    int ct = wave - rt * CT2;
    int sub  = lane & 15;
    int half = lane >> 4;

    // A fragment rows: rt*64 + 16*i + sub; half 0 -> K {0..7,16..23}, half 1 -> {8..15,24..31}
    const __bf16* ar[4];
#pragma unroll
    for (int i = 0; i < 4; ++i)
        ar[i] = A + (size_t)(rt * 64 + i * 16 + sub) * KP + half * 8;
    // B fragment cols: ct*32 + 16*j + sub; half 0 -> K {0..7,16..23}, half 1 -> {8..15,24..31}
    const __bf16* br[2];
#pragma unroll
    for (int j = 0; j < 2; ++j)
        br[j] = B + (size_t)(ct * 32 + j * 16 + sub) * KP + half * 8;

    v8f acc[4][2];
#pragma unroll
    for (int i = 0; i < 4; ++i)
#pragma unroll
        for (int j = 0; j < 2; ++j)
            acc[i][j] = {};

    for (int k0 = 0; k0 < KP; k0 += 32) {
        v16bf av[4], bv[2];
#pragma unroll
        for (int i = 0; i < 4; ++i) {
            v8bf a0 = *(const v8bf*)(ar[i] + k0);
            v8bf a1 = *(const v8bf*)(ar[i] + k0 + 16);
#pragma unroll
            for (int t = 0; t < 8; ++t) { av[i][t] = a0[t]; av[i][t + 8] = a1[t]; }
        }
#pragma unroll
        for (int j = 0; j < 2; ++j) {
            v8bf b0 = *(const v8bf*)(br[j] + k0);
            v8bf b1 = *(const v8bf*)(br[j] + k0 + 16);
#pragma unroll
            for (int t = 0; t < 8; ++t) { bv[j][t] = b0[t]; bv[j][t + 8] = b1[t]; }
        }
#pragma unroll
        for (int i = 0; i < 4; ++i)
#pragma unroll
            for (int j = 0; j < 2; ++j)
                acc[i][j] = __builtin_amdgcn_wmma_f32_16x16x32_bf16(
                                false, av[i], false, bv[j], (short)0, acc[i][j],
                                false, false);
    }

#pragma unroll
    for (int j = 0; j < 2; ++j) {
        int col = ct * 32 + j * 16 + sub;
        if (col < O) {
            float bb = (mode >= 1 && bias) ? bias[col] : 0.0f;
#pragma unroll
            for (int i = 0; i < 4; ++i) {
#pragma unroll
                for (int v = 0; v < 8; ++v) {
                    int row = rt * 64 + i * 16 + half * 8 + v;
                    float x = acc[i][j][v] + bb;
                    if (mode == 2 && x < 0.0f) x *= slope;
                    if (row < NN) C[(size_t)row * ldc + col] = x;
                }
            }
        }
    }
}

// f32 -> bf16 convert with zero padding: dst[DR,DC] from src[SR,SC]
__global__ __launch_bounds__(256)
void k_convert(const float* __restrict__ src, __bf16* __restrict__ dst,
               int SR, int SC, int DR, int DC)
{
    unsigned idx = blockIdx.x * 256u + threadIdx.x;
    unsigned tot = (unsigned)DR * (unsigned)DC;
    if (idx >= tot) return;
    int r = idx / DC, c = idx - r * DC;
    float v = (r < SR && c < SC) ? src[(size_t)r * SC + c] : 0.0f;
    dst[idx] = (__bf16)v;
}

// dst[n, ca+cb+cc] = concat(a,b,c) row-wise (c may be null with cc==0)
__global__ __launch_bounds__(256)
void k_concat3(float* __restrict__ dst, int ldd,
               const float* __restrict__ a, int ca,
               const float* __restrict__ b, int cb,
               const float* __restrict__ c, int cc, int n)
{
    int i = blockIdx.x;
    if (i >= n) return;
    for (int col = threadIdx.x; col < ldd; col += 256) {
        float v = 0.0f;
        if (col < ca)               v = a[(size_t)i * ca + col];
        else if (col < ca + cb)     v = b[(size_t)i * cb + (col - ca)];
        else if (c && col < ca + cb + cc) v = c[(size_t)i * cc + (col - ca - cb)];
        dst[(size_t)i * ldd + col] = v;
    }
}

__global__ __launch_bounds__(256)
void k_fill_u32(unsigned* p, unsigned v, int n)
{
    int i = blockIdx.x * 256 + threadIdx.x;
    if (i < n) p[i] = v;
}

// -------------------- CSR by destination --------------------
__global__ __launch_bounds__(256)
void k_count(const int* __restrict__ dst, int* cnt, int* pos, int E)
{
    int e = blockIdx.x * 256 + threadIdx.x;
    if (e >= E) return;
    pos[e] = atomicAdd(&cnt[dst[e]], 1);
}

__global__ __launch_bounds__(256)
void k_scan(const int* __restrict__ cnt, int* __restrict__ offs, int n)
{
    __shared__ int buf[256];
    __shared__ int carry;
    if (threadIdx.x == 0) carry = 0;
    __syncthreads();
    for (int base = 0; base < n; base += 256) {
        int i = base + (int)threadIdx.x;
        int v = (i < n) ? cnt[i] : 0;
        buf[threadIdx.x] = v;
        __syncthreads();
        for (int off = 1; off < 256; off <<= 1) {
            int t = (threadIdx.x >= (unsigned)off) ? buf[threadIdx.x - off] : 0;
            __syncthreads();
            buf[threadIdx.x] += t;
            __syncthreads();
        }
        if (i < n) offs[i] = carry + buf[threadIdx.x] - v;
        __syncthreads();
        if (threadIdx.x == 255) carry += buf[255];
        __syncthreads();
    }
    if (threadIdx.x == 0) offs[n] = carry;
}

__global__ __launch_bounds__(256)
void k_scatter(const int* __restrict__ dst, const int* __restrict__ offs,
               const int* __restrict__ pos, int* __restrict__ eidx, int E)
{
    int e = blockIdx.x * 256 + threadIdx.x;
    if (e >= E) return;
    eidx[offs[dst[e]] + pos[e]] = e;
}

// -------------------- GAT pieces --------------------
// za[i] = dot(z[i], wa[0:F]); zb[i] = dot(z[i], wa[F:2F])  (one wave per row)
__global__ __launch_bounds__(256)
void k_rowdot2(const float* __restrict__ z, int ldz, const float* __restrict__ wa,
               int F, float* __restrict__ za, float* __restrict__ zb, int n)
{
    int row = blockIdx.x * 8 + (threadIdx.x >> 5);
    int lane = threadIdx.x & 31;
    if (row >= n) return;
    const float* zr = z + (size_t)row * ldz;
    float a = 0.0f, b = 0.0f;
    for (int c = lane; c < F; c += 32) {
        float t = zr[c];
        a += t * wa[c];
        b += t * wa[F + c];
    }
    for (int off = 16; off; off >>= 1) {
        a += __shfl_down(a, off, 32);
        b += __shfl_down(b, off, 32);
    }
    if (lane == 0) { za[row] = a; zb[row] = b; }
}

// rep_pad(e_w) @ Wa_ew collapses to 2 scalars: c0 = sum(waew[0:half]), c1 = sum(waew[half:2half])
__global__ __launch_bounds__(256)
void k_ew_coeff(const float* __restrict__ waew, int half, float* __restrict__ outc)
{
    __shared__ float red[256];
    float a = 0.0f, b = 0.0f;
    for (int i = threadIdx.x; i < half; i += 256) { a += waew[i]; b += waew[half + i]; }
    red[threadIdx.x] = a; __syncthreads();
    for (int s = 128; s; s >>= 1) { if (threadIdx.x < (unsigned)s) red[threadIdx.x] += red[threadIdx.x + s]; __syncthreads(); }
    if (threadIdx.x == 0) outc[0] = red[0];
    __syncthreads();
    red[threadIdx.x] = b; __syncthreads();
    for (int s = 128; s; s >>= 1) { if (threadIdx.x < (unsigned)s) red[threadIdx.x] += red[threadIdx.x + s]; __syncthreads(); }
    if (threadIdx.x == 0) outc[1] = red[0];
}

__global__ __launch_bounds__(256)
void k_edge_logit(const int* __restrict__ src, const int* __restrict__ dst,
                  const float* __restrict__ za, const float* __restrict__ zb,
                  const float* __restrict__ ew, const float* __restrict__ ewc,
                  float* __restrict__ elog, int E)
{
    int e = blockIdx.x * 256 + threadIdx.x;
    if (e >= E) return;
    float t = za[src[e]] + zb[dst[e]];
    if (ewc) t += ewc[0] * ew[2 * e] + ewc[1] * ew[2 * e + 1];
    elog[e] = (t < 0.0f) ? 0.01f * t : t;       // leaky_relu(0.01)
}

__global__ __launch_bounds__(256)
void k_node_ms(const int* __restrict__ offs, const int* __restrict__ eidx,
               const float* __restrict__ elog, float* __restrict__ m,
               float* __restrict__ s, int n)
{
    int i = blockIdx.x * 256 + threadIdx.x;
    if (i >= n) return;
    int b = offs[i], e = offs[i + 1];
    float mm = -3.4e38f;
    for (int j = b; j < e; ++j) mm = fmaxf(mm, elog[eidx[j]]);
    if (!(mm > -3.0e38f)) mm = 0.0f;            // isfinite fallback (deg==0)
    float ss = 0.0f;
    for (int j = b; j < e; ++j) ss += __expf(elog[eidx[j]] - mm);
    m[i] = mm; s[i] = ss;
}

__global__ __launch_bounds__(256)
void k_edge_alpha(const int* __restrict__ dst, const float* __restrict__ elog,
                  const float* __restrict__ m, const float* __restrict__ s,
                  float* __restrict__ alpha, int E)
{
    int e = blockIdx.x * 256 + threadIdx.x;
    if (e >= E) return;
    int d = dst[e];
    float sd = s[d];
    alpha[e] = (sd > 0.0f) ? __expf(elog[e] - m[d]) / sd : 0.0f;
}

// out[i, ocol0+col] = h + relu( deg>0 ? hs + sum_e alpha*z[src] : h )
__global__ __launch_bounds__(256)
void k_agg_out(const int* __restrict__ offs, const int* __restrict__ eidx,
               const int* __restrict__ srcIdx, const float* __restrict__ alpha,
               const float* __restrict__ z, int ldz,
               const float* __restrict__ hs, int ldhs,
               const float* __restrict__ hres, int ldres,
               float* __restrict__ out, int ldo, int ocol0, int F)
{
    __shared__ int   s_src[64];
    __shared__ float s_a[64];
    int i = blockIdx.x;
    int col = blockIdx.y * 256 + threadIdx.x;
    int b = offs[i], e = offs[i + 1], deg = e - b;
    float acc = 0.0f;
    for (int base = b; base < e; base += 64) {
        int cnt = min(64, e - base);
        __syncthreads();
        if ((int)threadIdx.x < cnt) {
            int ed = eidx[base + threadIdx.x];
            s_src[threadIdx.x] = srcIdx[ed];
            s_a[threadIdx.x] = alpha[ed];
        }
        __syncthreads();
        if (col < F)
            for (int j = 0; j < cnt; ++j)
                acc += s_a[j] * z[(size_t)s_src[j] * ldz + col];
    }
    if (col < F) {
        float h  = hres[(size_t)i * ldres + col];
        float hn = (deg > 0) ? hs[(size_t)i * ldhs + col] + acc : h;
        out[(size_t)i * ldo + ocol0 + col] = h + fmaxf(hn, 0.0f);
    }
}

__global__ __launch_bounds__(256)
void k_reparam(const float* __restrict__ mu, const float* __restrict__ lv,
               const float* __restrict__ eps, float* __restrict__ z, int total)
{
    int i = blockIdx.x * 256 + threadIdx.x;
    if (i >= total) return;
    z[i] = mu[i] + __expf(0.5f * lv[i]) * eps[i];
}

// ---------------------------------------------------------------------------
extern "C" void kernel_launch(void* const* d_in, const int* in_sizes, int n_in,
                              void* d_out, int out_size, void* d_ws, size_t ws_size,
                              hipStream_t stream)
{
    (void)in_sizes; (void)n_in; (void)out_size; (void)ws_size;
    const float* feats    = (const float*)d_in[0];
    const float* gt       = (const float*)d_in[1];
    const float* maps_emb = (const float*)d_in[2];
    const float* e_w      = (const float*)d_in[3];
    const float* epsb     = (const float*)d_in[4];
    const int*   srcI     = (const int*)d_in[5];
    const int*   dstI     = (const int*)d_in[6];
    const float* emb_W    = (const float*)d_in[7];
    const float* emb_b    = (const float*)d_in[8];
    const float* enc1_Ws  = (const float*)d_in[9];
    const float* enc1_Wf  = (const float*)d_in[10];
    const float* enc1_Wa  = (const float*)d_in[11];
    const float* enc2_Ws  = (const float*)d_in[12];
    const float* enc2_Wf  = (const float*)d_in[13];
    const float* enc2_Wa  = (const float*)d_in[14];
    const float* encmlp_W = (const float*)d_in[15];
    const float* encmlp_b = (const float*)d_in[16];
    const float* mu_W     = (const float*)d_in[17];
    const float* mu_b     = (const float*)d_in[18];
    const float* lv_W     = (const float*)d_in[19];
    const float* lv_b     = (const float*)d_in[20];
    const float* dec1_Ws  = (const float*)d_in[21];
    const float* dec1_Wf  = (const float*)d_in[22];
    const float* dec1_Wa  = (const float*)d_in[23];
    const float* dec2_Ws  = (const float*)d_in[24];
    const float* dec2_Wf  = (const float*)d_in[25];
    const float* dec2_Wa  = (const float*)d_in[26];
    const float* dec0_W   = (const float*)d_in[27];
    const float* dec0_b   = (const float*)d_in[28];
    const float* dec1o_W  = (const float*)d_in[29];
    const float* dec1o_b  = (const float*)d_in[30];
    float* outp = (float*)d_out;

    // ---- workspace bump allocator ----
    uint8_t* wp = (uint8_t*)d_ws;
    auto alloc = [&](size_t bytes) -> void* {
        void* r = (void*)wp;
        wp += (bytes + 255) & ~(size_t)255;
        return r;
    };
    int*   e_pos  = (int*)alloc(NE * 4);
    int*   e_idx  = (int*)alloc(NE * 4);
    int*   counts = (int*)alloc(NN * 4);
    int*   offs   = (int*)alloc((NN + 1) * 4);
    __bf16* featsb = (__bf16*)alloc((size_t)NNP * 32 * 2);
    __bf16* embWb  = (__bf16*)alloc(256 * 32 * 2);
    __bf16* e1WsB0 = (__bf16*)alloc(544 * 544 * 2);
    __bf16* e1WsB1 = (__bf16*)alloc(544 * 544 * 2);
    __bf16* e1WfB0 = (__bf16*)alloc(544 * 544 * 2);
    __bf16* e1WfB1 = (__bf16*)alloc(544 * 544 * 2);
    __bf16* e2WsB  = (__bf16*)alloc((size_t)1088 * 1088 * 2);
    __bf16* e2WfB  = (__bf16*)alloc((size_t)1088 * 1088 * 2);
    __bf16* emlpWb = (__bf16*)alloc((size_t)576 * 1120 * 2);
    __bf16* muWb   = (__bf16*)alloc(32 * 576 * 2);
    __bf16* lvWb   = (__bf16*)alloc(32 * 576 * 2);
    __bf16* d1WsB0 = (__bf16*)alloc(544 * 544 * 2);
    __bf16* d1WsB1 = (__bf16*)alloc(544 * 544 * 2);
    __bf16* d1WfB0 = (__bf16*)alloc(544 * 544 * 2);
    __bf16* d1WfB1 = (__bf16*)alloc(544 * 544 * 2);
    __bf16* d2WsB  = (__bf16*)alloc((size_t)1088 * 1088 * 2);
    __bf16* d2WfB  = (__bf16*)alloc((size_t)1088 * 1088 * 2);
    __bf16* d0Wb   = (__bf16*)alloc((size_t)544 * 1120 * 2);
    __bf16* d1oWb  = (__bf16*)alloc(32 * 544 * 2);
    float* h_emb = (float*)alloc((size_t)NN * 256 * 4);
    float* actA  = (float*)alloc((size_t)NN * 1152 * 4);
    float* actB  = (float*)alloc((size_t)NN * 1152 * 4);
    float* actC  = (float*)alloc((size_t)NN * 1152 * 4);
    float* hsB   = (float*)alloc((size_t)NN * 1152 * 4);
    float* zfB   = (float*)alloc((size_t)NN * 1152 * 4);
    __bf16* Bact = (__bf16*)alloc((size_t)NNP * 1152 * 2);
    float* hm    = (float*)alloc((size_t)NN * 576 * 4);
    float* muB   = (float*)alloc((size_t)NN * 32 * 4);
    float* lvB   = (float*)alloc((size_t)NN * 32 * 4);
    float* zlat  = (float*)alloc((size_t)NN * 32 * 4);
    float* za    = (float*)alloc(NN * 4);
    float* zb    = (float*)alloc(NN * 4);
    float* mbuf  = (float*)alloc(NN * 4);
    float* sbuf  = (float*)alloc(NN * 4);
    float* elog  = (float*)alloc(NE * 4);
    float* alph  = (float*)alloc(NE * 4);
    float* ewc   = (float*)alloc(256);

    const int EB = (NE + 255) / 256;
    const int NB = (NN + 255) / 256;

    auto conv = [&](const float* s, __bf16* d, int SR, int SC, int DR, int DC) {
        unsigned tot = (unsigned)DR * (unsigned)DC;
        k_convert<<<(tot + 255) / 256, 256, 0, stream>>>(s, d, SR, SC, DR, DC);
    };
    auto gemm = [&](const __bf16* A, const __bf16* B, const float* bias, float* C,
                    int KP, int O, int ldc, int mode, float slope) {
        int CT2 = (O + 31) / 32;
        int waves = (NNP / 64) * CT2;               // 157 row blocks of 64
        k_gemm_bf16<<<(waves + 7) / 8, 256, 0, stream>>>(A, B, bias, C, KP, O, ldc,
                                                         CT2, waves, mode, slope);
    };
    auto gat_head = [&](const __bf16* Ab, const float* hres, int F, int FP,
                        const __bf16* WsB, const __bf16* WfB, const float* Wa,
                        bool has_ew, float* out, int ldo, int ocol0) {
        gemm(Ab, WsB, nullptr, hsB, FP, F, F, 0, 0.0f);
        gemm(Ab, WfB, nullptr, zfB, FP, F, F, 0, 0.0f);
        k_rowdot2<<<(NN + 7) / 8, 256, 0, stream>>>(zfB, F, Wa, F, za, zb, NN);
        const float* ewcp = nullptr;
        if (has_ew) {
            k_ew_coeff<<<1, 256, 0, stream>>>(Wa + 2 * F, F / 2, ewc);
            ewcp = ewc;
        }
        k_edge_logit<<<EB, 256, 0, stream>>>(srcI, dstI, za, zb, e_w, ewcp, elog, NE);
        k_node_ms<<<NB, 256, 0, stream>>>(offs, e_idx, elog, mbuf, sbuf, NN);
        k_edge_alpha<<<EB, 256, 0, stream>>>(dstI, elog, mbuf, sbuf, alph, NE);
        dim3 g(NN, (F + 255) / 256);
        k_agg_out<<<g, 256, 0, stream>>>(offs, e_idx, srcI, alph, zfB, F,
                                         hsB, F, hres, F, out, ldo, ocol0, F);
    };

    // ---- CSR by destination (once) ----
    k_fill_u32<<<NB, 256, 0, stream>>>((unsigned*)counts, 0u, NN);
    k_count<<<EB, 256, 0, stream>>>(dstI, counts, e_pos, NE);
    k_scan<<<1, 256, 0, stream>>>(counts, offs, NN);
    k_scatter<<<EB, 256, 0, stream>>>(dstI, offs, e_pos, e_idx, NE);

    // ---- weight conversion (bf16, rows padded to mult of 32) ----
    conv(emb_W, embWb, 256, 24, 256, 32);
    conv(enc1_Ws,             e1WsB0, 536, 536, 544, 544);
    conv(enc1_Ws + 536 * 536, e1WsB1, 536, 536, 544, 544);
    conv(enc1_Wf,             e1WfB0, 536, 536, 544, 544);
    conv(enc1_Wf + 536 * 536, e1WfB1, 536, 536, 544, 544);
    conv(enc2_Ws, e2WsB, 1072, 1072, 1088, 1088);
    conv(enc2_Wf, e2WfB, 1072, 1072, 1088, 1088);
    conv(encmlp_W, emlpWb, 548, 1096, 576, 1120);
    conv(mu_W, muWb, 32, 548, 32, 576);
    conv(lv_W, lvWb, 32, 548, 32, 576);
    conv(dec1_Ws,             d1WsB0, 544, 544, 544, 544);
    conv(dec1_Ws + 544 * 544, d1WsB1, 544, 544, 544, 544);
    conv(dec1_Wf,             d1WfB0, 544, 544, 544, 544);
    conv(dec1_Wf + 544 * 544, d1WfB1, 544, 544, 544, 544);
    conv(dec2_Ws, d2WsB, 1088, 1088, 1088, 1088);
    conv(dec2_Wf, d2WfB, 1088, 1088, 1088, 1088);
    conv(dec0_W, d0Wb, 544, 1120, 544, 1120);
    conv(dec1o_W, d1oWb, 24, 544, 32, 544);

    // ---- embedding: h_emb = feats @ emb_W^T + emb_b ----
    conv(feats, featsb, NN, 24, NNP, 32);
    gemm(featsb, embWb, emb_b, h_emb, 32, 256, 256, 1, 0.0f);

    // ---- encoder GAT layer 1 (2 heads, F=536) ----
    k_concat3<<<NN, 256, 0, stream>>>(actA, 536, maps_emb, 256, h_emb, 256, gt, 24, NN);
    conv(actA, Bact, NN, 536, NNP, 544);
    gat_head(Bact, actA, 536, 544, e1WsB0, e1WfB0, enc1_Wa,            true, actB, 1072, 0);
    gat_head(Bact, actA, 536, 544, e1WsB1, e1WfB1, enc1_Wa + 3 * 536,  true, actB, 1072, 536);

    // ---- encoder GAT layer 2 (1 head, F=1072) ----
    conv(actB, Bact, NN, 1072, NNP, 1088);
    gat_head(Bact, actB, 1072, 1088, e2WsB, e2WfB, enc2_Wa, true, actC, 1072, 0);

    // ---- MLP encoder + reparameterize ----
    k_concat3<<<NN, 256, 0, stream>>>(actA, 1096, actC, 1072, gt, 24, nullptr, 0, NN);
    conv(actA, Bact, NN, 1096, NNP, 1120);
    gemm(Bact, emlpWb, encmlp_b, hm, 1120, 548, 548, 2, 0.01f);
    conv(hm, Bact, NN, 548, NNP, 576);
    gemm(Bact, muWb, mu_b, muB, 576, 32, 32, 1, 0.0f);
    gemm(Bact, lvWb, lv_b, lvB, 576, 32, 32, 1, 0.0f);
    k_reparam<<<(NN * 32 + 255) / 256, 256, 0, stream>>>(muB, lvB, epsb, zlat, NN * 32);

    // ---- decoder GAT layer 1 (2 heads, F=544, no edge weights) ----
    k_concat3<<<NN, 256, 0, stream>>>(actA, 544, maps_emb, 256, h_emb, 256, zlat, 32, NN);
    conv(actA, Bact, NN, 544, NNP, 544);
    gat_head(Bact, actA, 544, 544, d1WsB0, d1WfB0, dec1_Wa,           false, actB, 1088, 0);
    gat_head(Bact, actA, 544, 544, d1WsB1, d1WfB1, dec1_Wa + 2 * 544, false, actB, 1088, 544);

    // ---- decoder GAT layer 2 (1 head, F=1088) ----
    conv(actB, Bact, NN, 1088, NNP, 1088);
    gat_head(Bact, actB, 1088, 1088, d2WsB, d2WfB, dec2_Wa, false, actC, 1088, 0);

    // ---- MLP decoder ----
    k_concat3<<<NN, 256, 0, stream>>>(actA, 1120, actC, 1088, zlat, 32, nullptr, 0, NN);
    conv(actA, Bact, NN, 1120, NNP, 1120);
    gemm(Bact, d0Wb, dec0_b, hm, 1120, 544, 544, 2, 0.02f);
    conv(hm, Bact, NN, 544, NNP, 544);
    gemm(Bact, d1oWb, dec1o_b, outp, 544, 24, 24, 1, 0.0f);
}
